// RTLTransformerBlock_90537910600206
// MI455X (gfx1250) — matrix operand order repaired and checked by reading, not compiled, not run on hardware
//
#include <hip/hip_runtime.h>
#include <math.h>
#include <stdint.h>

typedef __attribute__((ext_vector_type(16))) _Float16 v16h;
typedef __attribute__((ext_vector_type(8)))  _Float16 v8h;
typedef __attribute__((ext_vector_type(8)))  float    v8f;
typedef __attribute__((ext_vector_type(4)))  unsigned u32x4;
typedef __attribute__((ext_vector_type(8)))  unsigned u32x8;

#define TB_DIM   1024
#define TB_HEADS 16
#define TB_HD    64
#define TB_FF    4096
#define TB_B     2
#define TB_S     2048
#define TB_NTOK  (TB_B*TB_S)     // 4096
#define TB_EPS   1e-5f

#define TB_USE_TDM 1             // Tensor Data Mover staging for GEMM tiles

// ---------------------------------------------------------------- utilities
__global__ void cvt_f16_kernel(const float* __restrict__ in,
                               _Float16* __restrict__ out, int n) {
  int i = blockIdx.x * 256 + threadIdx.x;
  if (i < n) out[i] = (_Float16)in[i];
}

// layernorm over rows of [R, 1024] f32 -> f16
__global__ __launch_bounds__(256) void layernorm_kernel(
    const float* __restrict__ x, const float* __restrict__ g,
    const float* __restrict__ b, _Float16* __restrict__ out) {
  int row = blockIdx.x;
  const float* xr = x + (size_t)row * TB_DIM;
  int t = threadIdx.x;
  float v[4], s = 0.f, s2 = 0.f;
#pragma unroll
  for (int i = 0; i < 4; i++) { v[i] = xr[t + 256 * i]; s += v[i]; s2 += v[i] * v[i]; }
#pragma unroll
  for (int m = 16; m; m >>= 1) { s += __shfl_xor(s, m); s2 += __shfl_xor(s2, m); }
  __shared__ float ss[8], ss2[8];
  int w = t >> 5, l = t & 31;
  if (l == 0) { ss[w] = s; ss2[w] = s2; }
  __syncthreads();
  float S = 0.f, S2 = 0.f;
#pragma unroll
  for (int i = 0; i < 8; i++) { S += ss[i]; S2 += ss2[i]; }
  float mu = S * (1.f / TB_DIM);
  float var = S2 * (1.f / TB_DIM) - mu * mu;
  float r = rsqrtf(var + TB_EPS);
#pragma unroll
  for (int i = 0; i < 4; i++) {
    int c = t + 256 * i;
    out[(size_t)row * TB_DIM + c] = (_Float16)((v[i] - mu) * r * g[c] + b[c]);
  }
}

// ------------------------------------------------------- TDM descriptor load
// 2-D tile DMA: global (row-major, elem=f16) -> LDS, compacted row-major.
// D# layout per CDNA5 ISA 8.3/8.4 (group0: count/lds/global/type,
// group1: data_size, tensor dims, tile dims, stride).
__device__ __forceinline__ void tdm_load_tile_f16(
    uint32_t lds_off, const void* gaddr, uint32_t tensor_d0, uint32_t tensor_d1,
    uint32_t stride0, uint32_t tile_d0, uint32_t tile_d1) {
  uint64_t ga = (uint64_t)gaddr;
  union { uint32_t u[4]; u32x4 v; } g0;
  g0.u[0] = 1u;                                            // count=1, user D#
  g0.u[1] = lds_off;                                       // lds_addr [63:32]
  g0.u[2] = (uint32_t)ga;                                  // global_addr[31:0]
  g0.u[3] = (uint32_t)((ga >> 32) & 0x01FFFFFFu) | (2u << 30); // addr[56:32] | type=2
  union { uint32_t u[8]; u32x8 v; } g1;
  g1.u[0] = (1u << 16);                                    // data_size=1 (2 bytes)
  g1.u[1] = (tensor_d0 & 0xFFFFu) << 16;                   // tensor_dim0[15:0]
  g1.u[2] = (tensor_d0 >> 16) | ((tensor_d1 & 0xFFFFu) << 16); // dim0 hi | dim1 lo
  g1.u[3] = (tensor_d1 >> 16) | (tile_d0 << 16);           // dim1 hi | tile_dim0
  g1.u[4] = tile_d1 & 0xFFFFu;                             // tile_dim1 (tile_dim2=0)
  g1.u[5] = stride0;                                       // tensor_dim0_stride lo
  g1.u[6] = 0u;                                            // stride hi | dim1_stride lo
  g1.u[7] = 0u;
  asm volatile("tensor_load_to_lds %0, %1" :: "s"(g0.v), "s"(g1.v) : "memory");
}

// ---------------------------------------------------------------- WMMA GEMM
// out[m,n] = sum_k A[m,k] * W[n,k] + bias[n], epilogue per `mode`:
//   0: Q f16 [B,H,S,64] scaled 1/8   1: K f16 [B,H,S,64]
//   2: V f16 transposed [B,H,64,S]   3: GELU -> f16 [M,N]   4: +aux -> f32 [M,N]
__global__ __launch_bounds__(256) void wmma_gemm_kernel(
    const _Float16* __restrict__ A,   // [M,K]
    const _Float16* __restrict__ W,   // [N,K]
    const float* __restrict__ bias,   // [N]
    void* __restrict__ out,
    const float* __restrict__ aux,
    int M, int N, int K, int mode) {
  __shared__ __align__(16) _Float16 Sh[2][2][128 * 32];   // [stage][A|B]
  int t = threadIdx.x;
  int lane = t & 31, wid = t >> 5;
  int waveM = wid >> 2, waveN = wid & 3;      // 2 x 4 wave grid
  int bm = blockIdx.y * 128, bn = blockIdx.x * 128;
  int hi = lane >> 4, r0 = lane & 15;

  v8f acc[4][2] = {};
  int nstage = K >> 5;

#if TB_USE_TDM
  // stage 0: wave 0 DMAs the A tile, wave 1 the W tile
  if (wid == 0) {
    tdm_load_tile_f16((uint32_t)(size_t)(const void*)&Sh[0][0][0],
                      (const void*)(A + (size_t)bm * K), K, M, K, 32, 128);
  } else if (wid == 1) {
    tdm_load_tile_f16((uint32_t)(size_t)(const void*)&Sh[0][1][0],
                      (const void*)(W + (size_t)bn * K), K, N, K, 32, 128);
  }
#else
  int lrow = t >> 1, lcol = (t & 1) * 16;
#endif

  for (int st = 0; st < nstage; ++st) {
    int cur = st & 1;
#if TB_USE_TDM
    __builtin_amdgcn_s_wait_tensorcnt(0);     // stage `st` resident in LDS
    __syncthreads();                          // visible to all waves; prior reads done
    int kb = (st + 1) << 5;
    if (kb < K) {                             // prefetch next stage into other buffer
      if (wid == 0) {
        tdm_load_tile_f16((uint32_t)(size_t)(const void*)&Sh[cur ^ 1][0][0],
                          (const void*)(A + (size_t)bm * K + kb), K, M, K, 32, 128);
      } else if (wid == 1) {
        tdm_load_tile_f16((uint32_t)(size_t)(const void*)&Sh[cur ^ 1][1][0],
                          (const void*)(W + (size_t)bn * K + kb), K, N, K, 32, 128);
      }
    }
#else
    {
      int kb0 = st << 5;
      const _Float16* ap = A + (size_t)(bm + lrow) * K + kb0 + lcol;
      *(v8h*)&Sh[cur][0][lrow * 32 + lcol]     = *(const v8h*)ap;
      *(v8h*)&Sh[cur][0][lrow * 32 + lcol + 8] = *(const v8h*)(ap + 8);
      const _Float16* wp = W + (size_t)(bn + lrow) * K + kb0 + lcol;
      *(v8h*)&Sh[cur][1][lrow * 32 + lcol]     = *(const v8h*)wp;
      *(v8h*)&Sh[cur][1][lrow * 32 + lcol + 8] = *(const v8h*)(wp + 8);
      __syncthreads();
    }
#endif
    const _Float16* As = &Sh[cur][0][0];
    const _Float16* Bs = &Sh[cur][1][0];

    v16h af[4];
#pragma unroll
    for (int i = 0; i < 4; i++) {      // A frag: lane = row, halfs {b..b+7, b+16..b+23}
      int row = waveM * 64 + i * 16 + r0;
      union { v16h v; v8h h2[2]; } u;
      u.h2[0] = *(const v8h*)&As[row * 32 + hi * 8];
      u.h2[1] = *(const v8h*)&As[row * 32 + hi * 8 + 16];
      af[i] = u.v;
    }
    v16h bf[2];
#pragma unroll
    for (int j = 0; j < 2; j++) {      // B frag: lane = out-feature col, halfs {b..b+7, b+16..b+23}
      int fr = waveN * 32 + j * 16 + r0;
      union { v16h v; v8h h2[2]; } u;
      u.h2[0] = *(const v8h*)&Bs[fr * 32 + hi * 8];
      u.h2[1] = *(const v8h*)&Bs[fr * 32 + hi * 8 + 16];
      bf[j] = u.v;
    }
#pragma unroll
    for (int i = 0; i < 4; i++)
#pragma unroll
      for (int j = 0; j < 2; j++)
        acc[i][j] = __builtin_amdgcn_wmma_f32_16x16x32_f16(
            false, af[i], false, bf[j], (short)0, acc[i][j], false, false);
#if !TB_USE_TDM
    __syncthreads();
#endif
  }

  // epilogue: C layout — lane holds col n0, rows r + 8*hi
#pragma unroll
  for (int i = 0; i < 4; i++)
#pragma unroll
    for (int j = 0; j < 2; j++)
#pragma unroll
      for (int r = 0; r < 8; r++) {
        int m = r + 8 * hi;
        int rowg = bm + waveM * 64 + i * 16 + m;
        int colg = bn + waveN * 32 + j * 16 + r0;
        float val = acc[i][j][r] + bias[colg];
        if (mode == 0) {
          int h = colg >> 6, hd = colg & 63;
          int bb = rowg >> 11, s = rowg & 2047;
          ((_Float16*)out)[(((size_t)(bb * TB_HEADS + h) * TB_S + s) << 6) + hd] =
              (_Float16)(val * 0.125f);                     // 1/sqrt(64)
        } else if (mode == 1) {
          int h = colg >> 6, hd = colg & 63;
          int bb = rowg >> 11, s = rowg & 2047;
          ((_Float16*)out)[(((size_t)(bb * TB_HEADS + h) * TB_S + s) << 6) + hd] =
              (_Float16)val;
        } else if (mode == 2) {
          int h = colg >> 6, hd = colg & 63;
          int bb = rowg >> 11, s = rowg & 2047;
          ((_Float16*)out)[((size_t)((bb * TB_HEADS + h) * TB_HD + hd) * TB_S) + s] =
              (_Float16)val;
        } else if (mode == 3) {
          float gg = 0.5f * val * (1.0f + erff(val * 0.70710678118654752f));
          ((_Float16*)out)[(size_t)rowg * N + colg] = (_Float16)gg;
        } else {
          ((float*)out)[(size_t)rowg * N + colg] = val + aux[(size_t)rowg * N + colg];
        }
      }
}

// ---------------------------------------------------------------- attention
// one 16-row q tile per wave, flash-style online softmax with distance bias
__global__ __launch_bounds__(256) void flash_attn_kernel(
    const _Float16* __restrict__ Q,    // [BH,S,64] pre-scaled
    const _Float16* __restrict__ Km,   // [BH,S,64]
    const _Float16* __restrict__ Vt,   // [BH,64,S]
    const float* __restrict__ dscale, const float* __restrict__ dsens,
    const float* __restrict__ disso,
    _Float16* __restrict__ attn) {     // [NTOK, DIM]
  __shared__ __align__(16) _Float16 Pl[8][16 * 32];   // per-wave P staging
  int t = threadIdx.x, lane = t & 31, wid = t >> 5;
  int gw = blockIdx.x * 8 + wid;
  int qt = gw & 127;                   // S/16 = 128 q tiles
  int bh = gw >> 7;
  int h = bh & (TB_HEADS - 1);
  float pen = fabsf(dscale[h] * (1.0f - disso[0] * dsens[h]));
  int hi = lane >> 4, r0 = lane & 15;

  const _Float16* qp = Q + ((size_t)bh * TB_S + qt * 16 + r0) * TB_HD;
  v16h qa[2];
#pragma unroll
  for (int kk = 0; kk < 2; kk++) {
    int b0 = kk * 32 + hi * 8;
    union { v16h v; v8h h2[2]; } u;
    u.h2[0] = *(const v8h*)(qp + b0);
    u.h2[1] = *(const v8h*)(qp + b0 + 16);
    qa[kk] = u.v;
  }

  v8f o[4] = {};
  float mrow[8], lrow[8];
#pragma unroll
  for (int r = 0; r < 8; r++) { mrow[r] = -3.0e38f; lrow[r] = 0.f; }

  for (int kb = 0; kb < TB_S; kb += 32) {
    v8f sc[2];
#pragma unroll
    for (int j = 0; j < 2; j++) {
      v8f s = {};
#pragma unroll
      for (int kk = 0; kk < 2; kk++) {
        int kpos = kb + j * 16 + r0;
        const _Float16* kp = Km + ((size_t)bh * TB_S + kpos) * TB_HD + kk * 32 + hi * 8;
        union { v16h v; v8h h2[2]; } u;
        u.h2[0] = *(const v8h*)kp;
        u.h2[1] = *(const v8h*)(kp + 16);
        s = __builtin_amdgcn_wmma_f32_16x16x32_f16(
            false, qa[kk], false, u.v, (short)0, s, false, false);
      }
      sc[j] = s;
    }
    float alpha[8];
#pragma unroll
    for (int r = 0; r < 8; r++) {
      int qpos = qt * 16 + r + 8 * hi;
      float a0 = sc[0][r] - pen * fabsf((float)(qpos - (kb + r0)));
      float a1 = sc[1][r] - pen * fabsf((float)(qpos - (kb + 16 + r0)));
      float mx = fmaxf(a0, a1);
#pragma unroll
      for (int msk = 8; msk; msk >>= 1) mx = fmaxf(mx, __shfl_xor(mx, msk));
      float mnew = fmaxf(mrow[r], mx);
      alpha[r] = __expf(mrow[r] - mnew);
      float p0 = __expf(a0 - mnew);
      float p1 = __expf(a1 - mnew);
      float ps = p0 + p1;
#pragma unroll
      for (int msk = 8; msk; msk >>= 1) ps += __shfl_xor(ps, msk);
      lrow[r] = lrow[r] * alpha[r] + ps;
      mrow[r] = mnew;
      Pl[wid][(r + 8 * hi) * 32 + r0]      = (_Float16)p0;
      Pl[wid][(r + 8 * hi) * 32 + 16 + r0] = (_Float16)p1;
    }
#pragma unroll
    for (int tt = 0; tt < 4; tt++)
#pragma unroll
      for (int r = 0; r < 8; r++) o[tt][r] *= alpha[r];
    __syncthreads();
    union { v16h v; v8h h2[2]; } up;
    {
      const _Float16* pp = &Pl[wid][r0 * 32 + hi * 8];
      up.h2[0] = *(const v8h*)pp;
      up.h2[1] = *(const v8h*)(pp + 16);
    }
#pragma unroll
    for (int tt = 0; tt < 4; tt++) {
      const _Float16* vp = Vt + ((size_t)bh * TB_HD + tt * 16 + r0) * TB_S + kb + hi * 8;
      union { v16h v; v8h h2[2]; } uv;
      uv.h2[0] = *(const v8h*)vp;
      uv.h2[1] = *(const v8h*)(vp + 16);
      o[tt] = __builtin_amdgcn_wmma_f32_16x16x32_f16(
          false, up.v, false, uv.v, (short)0, o[tt], false, false);
    }
    __syncthreads();
  }

  int bb = bh >> 4;
#pragma unroll
  for (int tt = 0; tt < 4; tt++)
#pragma unroll
    for (int r = 0; r < 8; r++) {
      int s = qt * 16 + r + 8 * hi;
      int col = h * TB_HD + tt * 16 + r0;
      float v = o[tt][r] / fmaxf(lrow[r], 1e-20f);
      attn[(size_t)(bb * TB_S + s) * TB_DIM + col] = (_Float16)v;
    }
}

// ---------------------------------------------------------------- launcher
extern "C" void kernel_launch(void* const* d_in, const int* in_sizes, int n_in,
                              void* d_out, int out_size, void* d_ws, size_t ws_size,
                              hipStream_t stream) {
  const float* x      = (const float*)d_in[0];
  const float* disso  = (const float*)d_in[1];
  const float* wq = (const float*)d_in[2];  const float* bq = (const float*)d_in[3];
  const float* wk = (const float*)d_in[4];  const float* bk = (const float*)d_in[5];
  const float* wv = (const float*)d_in[6];  const float* bv = (const float*)d_in[7];
  const float* wo = (const float*)d_in[8];  const float* bo = (const float*)d_in[9];
  const float* dscale = (const float*)d_in[10];
  const float* dsens  = (const float*)d_in[11];
  const float* ln1g = (const float*)d_in[12]; const float* ln1b = (const float*)d_in[13];
  const float* ln2g = (const float*)d_in[14]; const float* ln2b = (const float*)d_in[15];
  const float* w1 = (const float*)d_in[16]; const float* b1 = (const float*)d_in[17];
  const float* w2 = (const float*)d_in[18]; const float* b2 = (const float*)d_in[19];
  float* out = (float*)d_out;

  char* wsb = (char*)d_ws; size_t off = 0;
  auto alloc = [&](size_t bytes) -> void* {
    void* p = wsb + off; off += (bytes + 255) & ~(size_t)255; return p;
  };
  _Float16* nrm    = (_Float16*)alloc((size_t)TB_NTOK * TB_DIM * 2);
  _Float16* q16    = (_Float16*)alloc((size_t)TB_NTOK * TB_DIM * 2);
  _Float16* k16    = (_Float16*)alloc((size_t)TB_NTOK * TB_DIM * 2);
  _Float16* vt16   = (_Float16*)alloc((size_t)TB_NTOK * TB_DIM * 2);
  _Float16* attn16 = (_Float16*)alloc((size_t)TB_NTOK * TB_DIM * 2);
  float*    x1     = (float*)   alloc((size_t)TB_NTOK * TB_DIM * 4);
  _Float16* h16    = (_Float16*)alloc((size_t)TB_NTOK * TB_DIM * 2);
  _Float16* ff16   = (_Float16*)alloc((size_t)TB_NTOK * TB_FF * 2);
  _Float16* wq16   = (_Float16*)alloc((size_t)TB_DIM * TB_DIM * 2);
  _Float16* wk16   = (_Float16*)alloc((size_t)TB_DIM * TB_DIM * 2);
  _Float16* wv16   = (_Float16*)alloc((size_t)TB_DIM * TB_DIM * 2);
  _Float16* wo16   = (_Float16*)alloc((size_t)TB_DIM * TB_DIM * 2);
  _Float16* w116   = (_Float16*)alloc((size_t)TB_FF * TB_DIM * 2);
  _Float16* w216   = (_Float16*)alloc((size_t)TB_DIM * TB_FF * 2);

  int nw = TB_DIM * TB_DIM;
  int nf = TB_FF * TB_DIM;
  cvt_f16_kernel<<<(nw + 255) / 256, 256, 0, stream>>>(wq, wq16, nw);
  cvt_f16_kernel<<<(nw + 255) / 256, 256, 0, stream>>>(wk, wk16, nw);
  cvt_f16_kernel<<<(nw + 255) / 256, 256, 0, stream>>>(wv, wv16, nw);
  cvt_f16_kernel<<<(nw + 255) / 256, 256, 0, stream>>>(wo, wo16, nw);
  cvt_f16_kernel<<<(nf + 255) / 256, 256, 0, stream>>>(w1, w116, nf);
  cvt_f16_kernel<<<(nf + 255) / 256, 256, 0, stream>>>(w2, w216, nf);

  layernorm_kernel<<<TB_NTOK, 256, 0, stream>>>(x, ln1g, ln1b, nrm);

  dim3 g1(TB_DIM / 128, TB_NTOK / 128);    // (8, 32)
  wmma_gemm_kernel<<<g1, 256, 0, stream>>>(nrm, wq16, bq, q16,  nullptr, TB_NTOK, TB_DIM, TB_DIM, 0);
  wmma_gemm_kernel<<<g1, 256, 0, stream>>>(nrm, wk16, bk, k16,  nullptr, TB_NTOK, TB_DIM, TB_DIM, 1);
  wmma_gemm_kernel<<<g1, 256, 0, stream>>>(nrm, wv16, bv, vt16, nullptr, TB_NTOK, TB_DIM, TB_DIM, 2);

  flash_attn_kernel<<<TB_B * TB_HEADS * (TB_S / 16) / 8, 256, 0, stream>>>(
      q16, k16, vt16, dscale, dsens, disso, attn16);

  wmma_gemm_kernel<<<g1, 256, 0, stream>>>(attn16, wo16, bo, x1, x, TB_NTOK, TB_DIM, TB_DIM, 4);

  layernorm_kernel<<<TB_NTOK, 256, 0, stream>>>(x1, ln2g, ln2b, h16);

  dim3 g2(TB_FF / 128, TB_NTOK / 128);     // (32, 32)
  wmma_gemm_kernel<<<g2, 256, 0, stream>>>(h16, w116, b1, ff16, nullptr, TB_NTOK, TB_FF, TB_DIM, 3);
  wmma_gemm_kernel<<<g1, 256, 0, stream>>>(ff16, w216, b2, out, x1, TB_NTOK, TB_DIM, TB_FF, 4);
}
